// PointNetFeaturePropagation_71210557768319
// MI455X (gfx1250) — matrix operand order repaired and checked by reading, not compiled, not run on hardware
//
#include <hip/hip_runtime.h>
#include <hip/hip_bf16.h>

typedef __bf16 bf16_t;
typedef __attribute__((ext_vector_type(16))) __bf16 v16bf;
typedef __attribute__((ext_vector_type(8)))  __bf16 v8bf;
typedef __attribute__((ext_vector_type(8)))  float  v8f;

#define B_  8
#define N_  8192
#define S_  2048
#define D1_ 64
#define D2_ 128
#define C0_ 192     // IN_CH
#define C1_ 256
#define C2_ 128
#define M_  (B_*N_) // 65536 rows

// ---------------------------------------------------------------------------
// Convert + transpose conv weights fp32 -> bf16, stored [out_ch][K]
// ---------------------------------------------------------------------------
__global__ __launch_bounds__(256) void cvt_w(const float* __restrict__ w0,
                                             const float* __restrict__ w1,
                                             bf16_t* __restrict__ w0t,
                                             bf16_t* __restrict__ w1t) {
  int tid = blockIdx.x * 256 + threadIdx.x;
  if (tid < C0_ * C1_) {              // w0: [192][256] -> w0t [256][192]
    int k = tid / C1_, n = tid % C1_;
    w0t[n * C0_ + k] = (bf16_t)w0[tid];
  }
  int tid2 = tid - C0_ * C1_;
  if (tid2 >= 0 && tid2 < C1_ * C2_) { // w1: [256][128] -> w1t [128][256]
    int k = tid2 / C2_, n = tid2 % C2_;
    w1t[n * C1_ + k] = (bf16_t)w1[tid2];
  }
}

// ---------------------------------------------------------------------------
// Brute-force 3-NN: one thread per query point, xyz2 of this batch in LDS.
// Branchless top-3 insertion network. Outputs indices + normalized weights.
// ---------------------------------------------------------------------------
__global__ __launch_bounds__(256) void knn_kernel(const float* __restrict__ xyz1,
                                                  const float* __restrict__ xyz2,
                                                  int*   __restrict__ oidx,
                                                  float* __restrict__ ow) {
  __shared__ float4 sp[S_];           // 32 KB
  int t  = threadIdx.x;
  int b  = blockIdx.x >> 5;           // 32 blocks of 256 pts per batch
  int n0 = (blockIdx.x & 31) << 8;
  const float* x2 = xyz2 + (size_t)b * 3 * S_;
  for (int s = t; s < S_; s += 256)
    sp[s] = make_float4(x2[s], x2[S_ + s], x2[2 * S_ + s], 0.f);
  __syncthreads();

  int n = n0 + t;
  const float* x1 = xyz1 + (size_t)b * 3 * N_;
  float px = x1[n], py = x1[N_ + n], pz = x1[2 * N_ + n];

  float d0 = 3.0e38f, d1 = 3.0e38f, d2 = 3.0e38f;
  int   i0 = 0, i1 = 0, i2 = 0;
  for (int s = 0; s < S_; ++s) {
    float4 q  = sp[s];
    float dx = px - q.x, dy = py - q.y, dz = pz - q.z;
    float d  = dx * dx + dy * dy + dz * dz;
    bool  c2 = d < d2;
    float nd = c2 ? d : d2;  int ni = c2 ? s : i2;
    bool  c1 = nd < d1;
    d2 = c1 ? d1 : nd;       i2 = c1 ? i1 : ni;
    float md = c1 ? nd : d1; int mi = c1 ? ni : i1;
    bool  c0 = md < d0;
    d1 = c0 ? d0 : md;       i1 = c0 ? i0 : mi;
    d0 = c0 ? md : d0;       i0 = c0 ? mi : i0;
  }
  float r0 = 1.f / (d0 + 1e-8f), r1 = 1.f / (d1 + 1e-8f), r2 = 1.f / (d2 + 1e-8f);
  float inv = 1.f / (r0 + r1 + r2);
  size_t row = (size_t)b * N_ + n;
  oidx[row * 3 + 0] = i0; oidx[row * 3 + 1] = i1; oidx[row * 3 + 2] = i2;
  ow[row * 3 + 0] = r0 * inv; ow[row * 3 + 1] = r1 * inv; ow[row * 3 + 2] = r2 * inv;
}

// ---------------------------------------------------------------------------
// Build h = [points1 | interp(points2)] row-major bf16 [M][192].
// One thread per row; writes 16B chunks. Gathers hit L2 (points2 = 8 MB).
// ---------------------------------------------------------------------------
__global__ __launch_bounds__(256) void build_h(const float* __restrict__ points1,
                                               const float* __restrict__ points2,
                                               const int*   __restrict__ kidx,
                                               const float* __restrict__ kw,
                                               bf16_t* __restrict__ hA) {
  size_t row = (size_t)blockIdx.x * 256 + threadIdx.x;
  int b = (int)(row >> 13);
  int n = (int)(row & (N_ - 1));
  const float* p1 = points1 + (size_t)b * D1_ * N_ + n;
  int   j0 = kidx[row * 3 + 0], j1 = kidx[row * 3 + 1], j2 = kidx[row * 3 + 2];
  float w0 = kw[row * 3 + 0],   w1 = kw[row * 3 + 1],   w2 = kw[row * 3 + 2];
  const float* p2 = points2 + (size_t)b * D2_ * S_;
  bf16_t* hrow = hA + row * C0_;

  #pragma unroll
  for (int c0 = 0; c0 < D1_; c0 += 8) {
    v8bf v;
    #pragma unroll
    for (int j = 0; j < 8; ++j) v[j] = (bf16_t)p1[(size_t)(c0 + j) * N_];
    *(v8bf*)(hrow + c0) = v;
  }
  #pragma unroll
  for (int c0 = 0; c0 < D2_; c0 += 8) {
    v8bf v;
    #pragma unroll
    for (int j = 0; j < 8; ++j) {
      const float* pc = p2 + (size_t)(c0 + j) * S_;
      v[j] = (bf16_t)(w0 * pc[j0] + w1 * pc[j1] + w2 * pc[j2]);
    }
    *(v8bf*)(hrow + D1_ + c0) = v;
  }
}

// ---------------------------------------------------------------------------
// bf16 GEMM via v_wmma_f32_16x16x32_bf16. A row-major [M][K], WT [Nc][K]
// (pre-transposed). Each wave owns a 32x32 tile (2 A-frags x 2 B-frags ->
// 4 WMMAs per k-step, 2 b128 loads per WMMA). Block = 8 waves arranged
// 4(M) x 2(N) -> 128x64 block tile.
// A-frag: lane(l) row = l%16; elems 0-7 = K 8h..8h+7, 8-15 = K 16+8h..  (ISA
// 16-bit A layout) -> two contiguous b128 loads. B-frag: same K order as A,
// col = l%16 -> two b128 loads from WT row.
// ---------------------------------------------------------------------------
template <typename OutT>
__global__ __launch_bounds__(256) void gemm_bf16(const bf16_t* __restrict__ A,
                                                 const bf16_t* __restrict__ WT,
                                                 const float*  __restrict__ bias,
                                                 OutT* __restrict__ out,
                                                 int K, int Nc) {
  int lane = threadIdx.x & 31;
  int wv   = threadIdx.x >> 5;       // 0..7
  int wm   = wv >> 1;                // 0..3 : M direction
  int wn   = wv & 1;                 // 0..1 : N direction
  int tilesN = Nc >> 6;              // 64-col block tiles
  int m0 = (blockIdx.x / tilesN) * 128 + wm * 32;
  int n0 = (blockIdx.x % tilesN) * 64 + wn * 32;
  int hf = lane >> 4, lm = lane & 15;

  const bf16_t* arow0 = A  + (size_t)(m0 + lm) * K;
  const bf16_t* arow1 = A  + (size_t)(m0 + 16 + lm) * K;
  const bf16_t* wrow0 = WT + (size_t)(n0 + lm) * K + 8 * hf;
  const bf16_t* wrow1 = WT + (size_t)(n0 + 16 + lm) * K + 8 * hf;

  v8f acc00 = {}, acc01 = {}, acc10 = {}, acc11 = {};
  for (int k0 = 0; k0 < K; k0 += 32) {
    union { v16bf v; v8bf h[2]; } a0, a1, b0, b1;
    a0.h[0] = *(const v8bf*)(arow0 + k0 + 8 * hf);
    a0.h[1] = *(const v8bf*)(arow0 + k0 + 16 + 8 * hf);
    a1.h[0] = *(const v8bf*)(arow1 + k0 + 8 * hf);
    a1.h[1] = *(const v8bf*)(arow1 + k0 + 16 + 8 * hf);
    b0.h[0] = *(const v8bf*)(wrow0 + k0);
    b0.h[1] = *(const v8bf*)(wrow0 + k0 + 16);
    b1.h[0] = *(const v8bf*)(wrow1 + k0);
    b1.h[1] = *(const v8bf*)(wrow1 + k0 + 16);
    acc00 = __builtin_amdgcn_wmma_f32_16x16x32_bf16(false, a0.v, false, b0.v,
                                                    (short)0, acc00, false, false);
    acc01 = __builtin_amdgcn_wmma_f32_16x16x32_bf16(false, a0.v, false, b1.v,
                                                    (short)0, acc01, false, false);
    acc10 = __builtin_amdgcn_wmma_f32_16x16x32_bf16(false, a1.v, false, b0.v,
                                                    (short)0, acc10, false, false);
    acc11 = __builtin_amdgcn_wmma_f32_16x16x32_bf16(false, a1.v, false, b1.v,
                                                    (short)0, acc11, false, false);
  }
  float bs0 = bias[n0 + lm];
  float bs1 = bias[n0 + 16 + lm];
  #pragma unroll
  for (int r = 0; r < 8; ++r) {
    size_t row0 = (size_t)(m0 + hf * 8 + r);        // C layout: M = r + 8*(lane/16)
    size_t row1 = row0 + 16;
    out[row0 * Nc + (n0 + lm)]      = (OutT)(acc00[r] + bs0);
    out[row0 * Nc + (n0 + 16 + lm)] = (OutT)(acc01[r] + bs1);
    out[row1 * Nc + (n0 + lm)]      = (OutT)(acc10[r] + bs0);
    out[row1 * Nc + (n0 + 16 + lm)] = (OutT)(acc11[r] + bs1);
  }
}

// ---------------------------------------------------------------------------
// Deterministic batch stats: stage A = per-block (256-row) partial sum/sumsq
// per channel; stage B = single block folds partials, emits (a, b) with
// a = scale*rsqrt(var+eps), b = bias - mean*a.
// ---------------------------------------------------------------------------
template <typename InT>
__global__ __launch_bounds__(256) void stats_partial(const InT* __restrict__ X,
                                                     float* __restrict__ partial,
                                                     int Nc) {
  int t    = threadIdx.x;
  int reps = 256 / Nc;                 // 1 (Nc=256) or 2 (Nc=128)
  int c    = t % Nc, rsub = t / Nc;
  int iters = 256 / reps;
  float s = 0.f, s2 = 0.f;
  size_t base = (size_t)blockIdx.x * 256;
  for (int i = 0; i < iters; ++i) {
    size_t row = base + (size_t)i * reps + rsub;
    float x = (float)X[row * Nc + c];
    s += x; s2 += x * x;
  }
  partial[(size_t)blockIdx.x * 512 + t]       = s;
  partial[(size_t)blockIdx.x * 512 + 256 + t] = s2;
}

__global__ __launch_bounds__(256) void stats_final(const float* __restrict__ partial,
                                                   const float* __restrict__ sc,
                                                   const float* __restrict__ bb,
                                                   float* __restrict__ ab,
                                                   int Nc, int nblocks) {
  int c = threadIdx.x;
  if (c >= Nc) return;
  int reps = 256 / Nc;
  float s = 0.f, s2 = 0.f;
  for (int blk = 0; blk < nblocks; ++blk)
    for (int r = 0; r < reps; ++r) {
      s  += partial[(size_t)blk * 512 + r * Nc + c];
      s2 += partial[(size_t)blk * 512 + 256 + r * Nc + c];
    }
  float invM = 1.f / (float)M_;
  float mu   = s * invM;
  float var  = s2 * invM - mu * mu;
  float a    = sc[c] * rsqrtf(var + 1e-5f);
  ab[c]       = a;
  ab[Nc + c]  = bb[c] - mu * a;
}

// BN + ReLU, in place on bf16 layer-1 activations (becomes GEMM2's A).
__global__ __launch_bounds__(256) void bn_relu_inplace(bf16_t* __restrict__ X,
                                                       const float* __restrict__ ab,
                                                       int Nc) {
  size_t idx = (size_t)blockIdx.x * 256 + threadIdx.x;
  int c = (int)(idx % Nc);
  float y = (float)X[idx] * ab[c] + ab[Nc + c];
  X[idx] = (bf16_t)(y > 0.f ? y : 0.f);
}

// BN + ReLU on layer-2 output + transpose [M][128] -> [B][128][N] fp32.
__global__ __launch_bounds__(256) void bn_relu_out(const float* __restrict__ T2,
                                                   const float* __restrict__ ab,
                                                   float* __restrict__ out) {
  size_t tid = (size_t)blockIdx.x * 256 + threadIdx.x; // = b*2^20 + c*2^13 + n
  int n = (int)(tid & (N_ - 1));
  int c = (int)((tid >> 13) & (C2_ - 1));
  int b = (int)(tid >> 20);
  size_t row = (size_t)b * N_ + n;
  float y = T2[row * C2_ + c] * ab[c] + ab[C2_ + c];
  out[tid] = y > 0.f ? y : 0.f;
}

// ---------------------------------------------------------------------------
extern "C" void kernel_launch(void* const* d_in, const int* in_sizes, int n_in,
                              void* d_out, int out_size, void* d_ws, size_t ws_size,
                              hipStream_t stream) {
  (void)in_sizes; (void)n_in; (void)out_size; (void)ws_size;
  const float* xyz1    = (const float*)d_in[0];
  const float* xyz2    = (const float*)d_in[1];
  const float* points1 = (const float*)d_in[2];
  const float* points2 = (const float*)d_in[3];
  const float* w0  = (const float*)d_in[4];
  const float* cb0 = (const float*)d_in[5];
  const float* s0  = (const float*)d_in[6];
  const float* b0  = (const float*)d_in[7];
  const float* w1  = (const float*)d_in[8];
  const float* cb1 = (const float*)d_in[9];
  const float* s1  = (const float*)d_in[10];
  const float* b1  = (const float*)d_in[11];
  float* out = (float*)d_out;

  char* base = (char*)d_ws;
  size_t off = 0;
  auto alloc = [&](size_t bytes) {
    void* p = base + off;
    off = (off + bytes + 255) & ~(size_t)255;
    return p;
  };
  bf16_t* w0t  = (bf16_t*)alloc((size_t)C1_ * C0_ * 2);
  bf16_t* w1t  = (bf16_t*)alloc((size_t)C2_ * C1_ * 2);
  int*    kidx = (int*)  alloc((size_t)M_ * 3 * 4);
  float*  kw   = (float*)alloc((size_t)M_ * 3 * 4);
  bf16_t* hA   = (bf16_t*)alloc((size_t)M_ * C0_ * 2);   // 24 MB
  bf16_t* t1   = (bf16_t*)alloc((size_t)M_ * C1_ * 2);   // 32 MB
  float*  t2   = (float*) alloc((size_t)M_ * C2_ * 4);   // 32 MB
  float*  part = (float*) alloc((size_t)256 * 512 * 4);  // stat partials
  float*  ab1  = (float*) alloc((size_t)2 * C1_ * 4);
  float*  ab2  = (float*) alloc((size_t)2 * C2_ * 4);

  cvt_w<<<dim3(320), dim3(256), 0, stream>>>(w0, w1, w0t, w1t);
  knn_kernel<<<dim3(B_ * 32), dim3(256), 0, stream>>>(xyz1, xyz2, kidx, kw);
  build_h<<<dim3(M_ / 256), dim3(256), 0, stream>>>(points1, points2, kidx, kw, hA);

  gemm_bf16<bf16_t><<<dim3((M_ / 128) * (C1_ / 64)), dim3(256), 0, stream>>>(
      hA, w0t, cb0, t1, C0_, C1_);
  stats_partial<bf16_t><<<dim3(256), dim3(256), 0, stream>>>(t1, part, C1_);
  stats_final<<<dim3(1), dim3(256), 0, stream>>>(part, s0, b0, ab1, C1_, 256);
  bn_relu_inplace<<<dim3((unsigned)((size_t)M_ * C1_ / 256)), dim3(256), 0, stream>>>(
      t1, ab1, C1_);

  gemm_bf16<float><<<dim3((M_ / 128) * (C2_ / 64)), dim3(256), 0, stream>>>(
      t1, w1t, cb1, t2, C1_, C2_);
  stats_partial<float><<<dim3(256), dim3(256), 0, stream>>>(t2, part, C2_);
  stats_final<<<dim3(1), dim3(256), 0, stream>>>(part, s1, b1, ab2, C2_, 256);
  bn_relu_out<<<dim3((unsigned)((size_t)B_ * C2_ * N_ / 256)), dim3(256), 0, stream>>>(
      t2, ab2, out);
}
